// CoAttention_5454608466044
// MI455X (gfx1250) — matrix operand order repaired and checked by reading, not compiled, not run on hardware
//
#include <hip/hip_runtime.h>
#include <hip/hip_bf16.h>

typedef __attribute__((ext_vector_type(16))) __bf16 v16bf;
typedef __attribute__((ext_vector_type(8)))  __bf16 v8bf;
typedef __attribute__((ext_vector_type(8)))  float  v8f;

#define WMMA_BF16(a, b, c) \
  __builtin_amdgcn_wmma_f32_16x16x32_bf16(false, (a), false, (b), (short)0, (c), false, false)

#if __has_builtin(__builtin_amdgcn_tensor_load_to_lds)
#define HAVE_TDM 1
typedef __attribute__((ext_vector_type(4))) unsigned int v4u;
typedef __attribute__((ext_vector_type(8))) int          v8i;
typedef __attribute__((ext_vector_type(4))) int          v4i;
#else
#define HAVE_TDM 0
#endif

static constexpr int  B_   = 8;
static constexpr int  LS   = 2048;
static constexpr int  LX   = 2048;
static constexpr int  DIM  = 512;

// ---------------------------------------------------------------------------
// f32 -> bf16 elementwise convert
// ---------------------------------------------------------------------------
__global__ void cvt_bf16_kernel(const float* __restrict__ in,
                                __bf16* __restrict__ out, size_t n) {
  size_t i = (size_t)blockIdx.x * blockDim.x + threadIdx.x;
  size_t stride = (size_t)gridDim.x * blockDim.x;
  for (; i < n; i += stride) out[i] = (__bf16)in[i];
}

// ---------------------------------------------------------------------------
// [B, L, D] f32 -> [B, D, L] bf16 tiled transpose
// ---------------------------------------------------------------------------
__global__ void transpose_cvt_kernel(const float* __restrict__ in,
                                     __bf16* __restrict__ out, int L, int D) {
  __shared__ float tile[32][33];
  int b = blockIdx.z;
  int x0 = blockIdx.x * 32, d0 = blockIdx.y * 32;
  int tx = threadIdx.x, ty = threadIdx.y;
  for (int j = ty; j < 32; j += 8)
    tile[j][tx] = in[((size_t)b * L + x0 + j) * D + d0 + tx];
  __syncthreads();
  for (int j = ty; j < 32; j += 8)
    out[((size_t)b * D + d0 + j) * L + x0 + tx] = (__bf16)tile[tx][j];
}

// ---------------------------------------------------------------------------
// proj[s, e] = sum_d seq[s, d] * W[e, d]   (bf16 WMMA, f32 accum -> bf16 out)
// grid.x = (B*LS)/16 row tiles, grid.y = DIM/64, 4 waves x 16 e-cols each
// ---------------------------------------------------------------------------
__global__ __launch_bounds__(128) void proj_kernel(const __bf16* __restrict__ X,
                                                   const __bf16* __restrict__ Wr,
                                                   __bf16* __restrict__ P, int D) {
  int tid = threadIdx.x;
  int lane = tid & 31;
  int wv = tid >> 5;
  int lrow = lane & 15, lhalf = lane >> 4;
  size_t s0 = (size_t)blockIdx.x * 16;
  int e0 = blockIdx.y * 64 + wv * 16;

  const __bf16* ab = X + (s0 + lrow) * D + lhalf * 8;           // A: seq rows
  const __bf16* bb = Wr + (size_t)(e0 + lrow) * D + lhalf * 8;  // B: W rows (B^T form)

  v8f acc = {};
#pragma unroll
  for (int k = 0; k < DIM; k += 32) {
    v8bf alo = *(const v8bf*)(ab + k);
    v8bf ahi = *(const v8bf*)(ab + k + 16);
    v16bf a = __builtin_shufflevector(alo, ahi, 0,1,2,3,4,5,6,7,8,9,10,11,12,13,14,15);
    v8bf blo = *(const v8bf*)(bb + k);
    v8bf bhi = *(const v8bf*)(bb + k + 16);
    v16bf b = __builtin_shufflevector(blo, bhi, 0,1,2,3,4,5,6,7,8,9,10,11,12,13,14,15);
    acc = WMMA_BF16(a, b, acc);
  }
#pragma unroll
  for (int r = 0; r < 8; ++r) {
    int M = r + lhalf * 8;
    P[(s0 + M) * D + e0 + lrow] = (__bf16)acc[r];
  }
}

// ---------------------------------------------------------------------------
// Fused co-attention direction:
//   S = Q K^T (16 q-rows x LK, exact strip in LDS), softmax over LK
//   (optionally masking K columns), Out = softmax(S) V   with V given as V^T.
// One workgroup = 16 query rows, 128 threads = 4 waves.
// ---------------------------------------------------------------------------
template <bool MASKED>
__global__ __launch_bounds__(128) void coattn_kernel(
    const __bf16* __restrict__ Q,   // [B, LQ, D]
    const __bf16* __restrict__ K,   // [B, LK, D]
    const __bf16* __restrict__ VT,  // [B, D, LK]
    const int* __restrict__ mask,   // [B, LK] (MASKED only)
    float* __restrict__ Out,        // [B, LQ, D]
    int LQ, int LK, int D) {
  extern __shared__ float smem[];
  const int SSTR  = LK + 4;          // padded f32 row stride (bank-spread)
  const int RED   = 16 * SSTR;       // 16x8 partial reductions
  const int RMAX  = RED + 128;       // 16 row maxima
  const int RINV  = RMAX + 16;       // 16 reciprocal row sums
  const int ABASE = RINV + 16;       // A strip (bf16, padded)
  const int ASTR  = D + 8;
  __bf16* aS = (__bf16*)(smem + ABASE);

  int tid = threadIdx.x;
  int lane = tid & 31;
  int wv = tid >> 5;
  int lrow = lane & 15, lhalf = lane >> 4;

  int qtiles = LQ >> 4;
  int b  = blockIdx.x / qtiles;
  int q0 = (blockIdx.x - b * qtiles) * 16;

  // ---- Phase 0: stage Q strip (16 x D bf16) into padded LDS -------------
#if HAVE_TDM
  // Tensor Data Mover: one descriptor DMAs the 16xD bf16 tile and applies the
  // +8 bf16 row pad in hardware (pad_interval = 256 dwords = one D=512 row,
  // pad_amount = 4 dwords = 8 bf16  ->  LDS row stride = ASTR = D + 8).
  if (wv == 0) {
    uint32_t lds_off = (uint32_t)(uintptr_t)aS;   // low 32 bits of flat = LDS byte offset
    uint64_t gaddr   = (uint64_t)(uintptr_t)(Q + ((size_t)b * LQ + q0) * D);
    v4u g0;
    g0[0] = 1u;                                            // count=1, user mode
    g0[1] = lds_off;                                       // lds_addr
    g0[2] = (uint32_t)gaddr;                               // global_addr[31:0]
    g0[3] = (uint32_t)((gaddr >> 32) & 0x01FFFFFFu) | (2u << 30);  // addr[56:32] | type=2
    v8i g1;
    g1[0] = (int)((1u << 16) | (1u << 20) | (7u << 22) | (3u << 25));
    //            data_size=2B | pad_en   | interval=256dw | amount=4dw ; wg_mask=0
    g1[1] = (int)((uint32_t)D << 16);    // tensor_dim0[15:0] in [31:16]
    g1[2] = (int)((uint32_t)LQ << 16);   // tensor_dim0 hi=0 | tensor_dim1 lo
    g1[3] = (int)((uint32_t)D << 16);    // tensor_dim1 hi=0 | tile_dim0 = D
    g1[4] = 16;                          // tile_dim1 = 16 rows, tile_dim2 = 0
    g1[5] = D;                           // tensor_dim0_stride lo
    g1[6] = 0;                           // stride0 hi | stride1 lo
    g1[7] = 0;                           // stride1 hi
    v4i gz = {0, 0, 0, 0};
#if defined(__clang_major__) && (__clang_major__ >= 23)
    v8i gz8 = {0, 0, 0, 0, 0, 0, 0, 0};
    __builtin_amdgcn_tensor_load_to_lds(g0, g1, gz, gz, gz8, 0);
#else
    __builtin_amdgcn_tensor_load_to_lds(g0, g1, gz, gz, 0);
#endif
    __builtin_amdgcn_s_wait_tensorcnt(0);
  }
#else
  {
    const uint32_t* src = (const uint32_t*)(Q + ((size_t)b * LQ + q0) * D);
    uint32_t* dst = (uint32_t*)aS;
    int nd = D >> 1, astr_dw = ASTR >> 1;
    for (int i = tid; i < 16 * nd; i += 128) {
      int r = i / nd, c = i - r * nd;
      dst[r * astr_dw + c] = src[r * nd + c];
    }
  }
#endif
  __syncthreads();

  // ---- Phase 1: S strip = Q * K^T  (each wave: LK/4 contiguous columns) -
  {
    const __bf16* ab = aS + lrow * ASTR + lhalf * 8;
    int ntiles = (LK >> 4) >> 2;
    for (int t = 0; t < ntiles; ++t) {
      int x0 = (wv * ntiles + t) * 16;
      if (t + 1 < ntiles) {   // prefetch next K-row tile (global_prefetch_b8)
        __builtin_prefetch((const void*)(K + ((size_t)b * LK + x0 + 16 + lrow) * D), 0, 3);
      }
      const __bf16* kb = K + ((size_t)b * LK + x0 + lrow) * D + lhalf * 8;
      v8f acc = {};
#pragma unroll
      for (int k = 0; k < DIM; k += 32) {
        v8bf alo = *(const v8bf*)(ab + k);
        v8bf ahi = *(const v8bf*)(ab + k + 16);
        v16bf a = __builtin_shufflevector(alo, ahi, 0,1,2,3,4,5,6,7,8,9,10,11,12,13,14,15);
        v8bf blo = *(const v8bf*)(kb + k);
        v8bf bhi = *(const v8bf*)(kb + k + 16);
        v16bf bm = __builtin_shufflevector(blo, bhi, 0,1,2,3,4,5,6,7,8,9,10,11,12,13,14,15);
        acc = WMMA_BF16(a, bm, acc);
      }
      bool keep = true;
      if (MASKED) keep = (mask[(size_t)b * LK + x0 + lrow] != 0);
      int col = x0 + lrow;  // C tile: N = lane%16
#pragma unroll
      for (int r = 0; r < 8; ++r) {
        int M = r + lhalf * 8;
        smem[M * SSTR + col] = keep ? acc[r] : -3.0e38f;
      }
    }
  }
  __syncthreads();

  // ---- Phase 2: exact per-row softmax stats (max, 1/sum) ----------------
  {
    int row = tid & 15, seg = tid >> 4;  // 16 rows x 8 segments
    int cpt = LK >> 3;
    const float* srow = smem + row * SSTR;
    float m = -3.4e38f;
    for (int c = seg * cpt; c < seg * cpt + cpt; ++c) m = fmaxf(m, srow[c]);
    smem[RED + row * 8 + seg] = m;
    __syncthreads();
    if (tid < 16) {
      float mm = -3.4e38f;
      for (int s = 0; s < 8; ++s) mm = fmaxf(mm, smem[RED + tid * 8 + s]);
      smem[RMAX + tid] = mm;
    }
    __syncthreads();
    float rm = smem[RMAX + row];
    float sum = 0.f;
    for (int c = seg * cpt; c < seg * cpt + cpt; ++c) sum += __expf(srow[c] - rm);
    smem[RED + row * 8 + seg] = sum;
    __syncthreads();
    if (tid < 16) {
      float ss = 0.f;
      for (int s = 0; s < 8; ++s) ss += smem[RED + tid * 8 + s];
      smem[RINV + tid] = 1.0f / ss;
    }
    __syncthreads();
  }

  // ---- Phase 3: Out = P * V  (each wave owns a D/4 = 128 column slab) ---
  {
    int d0 = wv * (D >> 2);
    float rm = smem[RMAX + lrow];
    v8f acc[8] = {};
    for (int x0 = 0; x0 < LK; x0 += 32) {
      const __bf16* vb = VT + ((size_t)b * D + d0 + lrow) * LK + x0 + lhalf * 8;
      if (x0 + 32 < LK) {   // prefetch next V^T chunk
        __builtin_prefetch((const void*)(vb + 32), 0, 3);
      }
      // Build P operand in A layout: lane row = lane%16,
      // K chunks [x0+h*8, +8) and [x0+16+h*8, +8)
      const float* sp = smem + lrow * SSTR + x0 + lhalf * 8;
      v16bf p;
#pragma unroll
      for (int e = 0; e < 8; ++e) {
        p[e]     = (__bf16)__expf(sp[e]      - rm);
        p[e + 8] = (__bf16)__expf(sp[e + 16] - rm);
      }
#pragma unroll
      for (int ns = 0; ns < 8; ++ns) {
        v8bf vlo = *(const v8bf*)(vb + (size_t)ns * 16 * LK);
        v8bf vhi = *(const v8bf*)(vb + (size_t)ns * 16 * LK + 16);
        v16bf bv = __builtin_shufflevector(vlo, vhi, 0,1,2,3,4,5,6,7,8,9,10,11,12,13,14,15);
        acc[ns] = WMMA_BF16(p, bv, acc[ns]);
      }
    }
    float* ob = Out + ((size_t)b * LQ + q0) * D + d0;
#pragma unroll
    for (int ns = 0; ns < 8; ++ns) {
#pragma unroll
      for (int r = 0; r < 8; ++r) {
        int M = r + lhalf * 8;
        ob[(size_t)M * D + ns * 16 + lrow] = acc[ns][r] * smem[RINV + M];
      }
    }
  }
}

// ---------------------------------------------------------------------------
extern "C" void kernel_launch(void* const* d_in, const int* in_sizes, int n_in,
                              void* d_out, int out_size, void* d_ws, size_t ws_size,
                              hipStream_t stream) {
  (void)in_sizes; (void)n_in; (void)out_size; (void)ws_size;

  const float* seq  = (const float*)d_in[0];   // [B, LS, D]
  const float* strf = (const float*)d_in[1];   // [B, LX, D]
  const int*   msk  = (const int*)d_in[2];     // [B, LS]
  const float* W    = (const float*)d_in[3];   // [D, D]

  float* att_seq    = (float*)d_out;                            // [B, LS, D]
  float* att_struct = (float*)d_out + (size_t)B_ * LS * DIM;    // [B, LX, D]

  const size_t NSEQ = (size_t)B_ * LS * DIM;
  const size_t NSTR = (size_t)B_ * LX * DIM;
  __bf16* seq_bf  = (__bf16*)d_ws;
  __bf16* str_bf  = seq_bf  + NSEQ;
  __bf16* strT_bf = str_bf  + NSTR;   // [B, D, LX]
  __bf16* seqT_bf = strT_bf + NSTR;   // [B, D, LS]
  __bf16* proj_bf = seqT_bf + NSEQ;   // [B, LS, D]
  __bf16* w_bf    = proj_bf + NSEQ;   // [D, D]

  cvt_bf16_kernel<<<2048, 256, 0, stream>>>(seq, seq_bf, NSEQ);
  cvt_bf16_kernel<<<2048, 256, 0, stream>>>(strf, str_bf, NSTR);
  cvt_bf16_kernel<<<512, 256, 0, stream>>>(W, w_bf, (size_t)DIM * DIM);

  dim3 tb(32, 8);
  transpose_cvt_kernel<<<dim3(LX / 32, DIM / 32, B_), tb, 0, stream>>>(strf, strT_bf, LX, DIM);
  transpose_cvt_kernel<<<dim3(LS / 32, DIM / 32, B_), tb, 0, stream>>>(seq, seqT_bf, LS, DIM);

  proj_kernel<<<dim3((B_ * LS) / 16, DIM / 64), 128, 0, stream>>>(seq_bf, w_bf, proj_bf, DIM);

  size_t shmem = (size_t)(16 * (LX + 4) + 160) * sizeof(float)   // S strip + stats
               + (size_t)16 * (DIM + 8) * sizeof(__bf16);        // padded A strip
  hipFuncSetAttribute(reinterpret_cast<const void*>(&coattn_kernel<false>),
                      hipFuncAttributeMaxDynamicSharedMemorySize, (int)shmem);
  hipFuncSetAttribute(reinterpret_cast<const void*>(&coattn_kernel<true>),
                      hipFuncAttributeMaxDynamicSharedMemorySize, (int)shmem);

  // seq -> struct: Q = proj rows, K = struct rows, V^T = structT, no mask
  coattn_kernel<false><<<B_ * (LS / 16), 128, shmem, stream>>>(
      proj_bf, str_bf, strT_bf, nullptr, att_seq, LS, LX, DIM);

  // struct -> seq: Q = struct rows, K = proj rows, V^T = seqT, mask on seq cols
  coattn_kernel<true><<<B_ * (LX / 16), 128, shmem, stream>>>(
      str_bf, proj_bf, seqT_bf, msk, att_struct, LX, LS, DIM);
}
